// POIGraph_88545045775037
// MI455X (gfx1250) — matrix operand order repaired and checked by reading, not compiled, not run on hardware
//
#include <hip/hip_runtime.h>
#include <hip/amd_detail/amd_hip_unsafe_atomics.h>

#define NNODES 50000
#define HID 512
#define NEDGES 200000
#define NLAYERS 2

typedef __bf16 bf16_t;
typedef __bf16 v16bf __attribute__((ext_vector_type(16)));
typedef __bf16 v8bf  __attribute__((ext_vector_type(8)));
typedef __bf16 v4bf  __attribute__((ext_vector_type(4)));
typedef float  v8f   __attribute__((ext_vector_type(8)));

// A-fragment assembled from two 8-element LDS chunks (contiguous K order:
// lanes 0-15 hold K = k0..k0+15, lanes 16-31 hold k0+16..k0+31; B uses the same order).
union FragA { v16bf v; v8bf h[2]; };

// D = A(16x32 bf16) * B(32x16 bf16) + C(f32); 8-arg VOP3P form.
__device__ __forceinline__ v8f wmma_bf16(v16bf a, v16bf b, v8f c) {
  return __builtin_amdgcn_wmma_f32_16x16x32_bf16(false, a, false, b, (short)0, c,
                                                 false, false);
}

__device__ __forceinline__ float sigmoidf_(float v) {
  return 1.0f / (1.0f + __expf(-v));
}

// ---------------------------------------------------------------- gather x = emb[inputs]
__global__ void ggnn_gather(const int* __restrict__ inputs,
                            const float* __restrict__ emb,
                            float* __restrict__ x) {
  int gid = blockIdx.x * blockDim.x + threadIdx.x;   // one float4 per thread
  int i = gid >> 7;
  int j = (gid & 127) << 2;
  if (i >= NNODES) return;
  int idx = inputs[i];
  float4 v = *(const float4*)(emb + (size_t)idx * HID + j);
  *(float4*)(x + (size_t)i * HID + j) = v;
}

// ---------------------------------------------------------------- zero fill
__global__ void ggnn_zero(float* __restrict__ p, int n4) {
  int gid = blockIdx.x * blockDim.x + threadIdx.x;
  if (gid >= n4) return;
  float4 z = {0.f, 0.f, 0.f, 0.f};
  *(float4*)(p + (size_t)gid * 4) = z;
}

// ---------------------------------------------------------------- agg[dst] += m[src]
__global__ void ggnn_scatter(const int* __restrict__ A,
                             const float* __restrict__ m,
                             float* __restrict__ agg) {
  int gid = blockIdx.x * blockDim.x + threadIdx.x;   // (edge, 4 floats)
  int e = gid >> 7;
  int j = (gid & 127) << 2;
  if (e >= NEDGES) return;
  int src = A[e];
  int dst = A[NEDGES + e];
  float4 v = *(const float4*)(m + (size_t)src * HID + j);
  float* p = agg + (size_t)dst * HID + j;
  unsafeAtomicAdd(p + 0, v.x);
  unsafeAtomicAdd(p + 1, v.y);
  unsafeAtomicAdd(p + 2, v.z);
  unsafeAtomicAdd(p + 3, v.w);
}

// ---------------------------------------------------------------- fp32 -> bf16 hi/lo split (flat)
__global__ void ggnn_split(const float* __restrict__ src,
                           bf16_t* __restrict__ hi, bf16_t* __restrict__ lo,
                           int n4) {
  int gid = blockIdx.x * blockDim.x + threadIdx.x;
  if (gid >= n4) return;
  float4 v = ((const float4*)src)[gid];
  float f[4] = {v.x, v.y, v.z, v.w};
  v4bf h, l;
#pragma unroll
  for (int q = 0; q < 4; ++q) {
    bf16_t hb = (bf16_t)f[q];
    h[q] = hb;
    l[q] = (bf16_t)(f[q] - (float)hb);
  }
  ((v4bf*)hi)[gid] = h;
  ((v4bf*)lo)[gid] = l;
}

// ---------------------------------------------------------------- W (KxN) -> Wt (NxK) hi/lo split
__global__ void ggnn_tsplit_w(const float* __restrict__ W,
                              bf16_t* __restrict__ hi, bf16_t* __restrict__ lo) {
  int gid = blockIdx.x * blockDim.x + threadIdx.x;   // gid = n*512 + k
  int n = gid >> 9;
  int k = gid & (HID - 1);
  float f = W[(size_t)k * HID + n];
  bf16_t hb = (bf16_t)f;
  hi[gid] = hb;
  lo[gid] = (bf16_t)(f - (float)hb);
}

// ---------------------------------------------------------------- m = x @ W (split bf16 WMMA)
// Block: 8 waves, one 16-row node tile. Wt is [N,K] row-major (K contiguous),
// so B fragments are 16 contiguous bf16 per lane.
__global__ __launch_bounds__(256) void ggnn_gemm_bf16(
    const float* __restrict__ x, const bf16_t* __restrict__ wth,
    const bf16_t* __restrict__ wtl, float* __restrict__ m) {
  __shared__ bf16_t sh[16 * HID];                    // 16 KB
  __shared__ bf16_t sl[16 * HID];                    // 16 KB
  const int row0 = blockIdx.x * 16;
  const int tid = threadIdx.x;

  // fused fp32 load + hi/lo split into LDS (tile is contiguous in memory)
  const float* src = x + (size_t)row0 * HID;
  for (int i = tid; i < 16 * HID / 4; i += 256) {
    float4 v = ((const float4*)src)[i];
    float f[4] = {v.x, v.y, v.z, v.w};
    v4bf h, l;
#pragma unroll
    for (int q = 0; q < 4; ++q) {
      bf16_t hb = (bf16_t)f[q];
      h[q] = hb;
      l[q] = (bf16_t)(f[q] - (float)hb);
    }
    *(v4bf*)(sh + i * 4) = h;
    *(v4bf*)(sl + i * 4) = l;
  }
  __syncthreads();

  const int wave = tid >> 5, lane = tid & 31;
  const int lrow = lane & 15;
  const int h8 = (lane >> 4) << 3;                   // 0 or 8
  const int chalf = (lane >> 4) << 3;

  for (int t = 0; t < 4; ++t) {
    const int ccol = wave * 64 + t * 16 + lrow;
    const bf16_t* bh = wth + (size_t)ccol * HID;
    const bf16_t* bl = wtl + (size_t)ccol * HID;
    v8f acc = {0.f, 0.f, 0.f, 0.f, 0.f, 0.f, 0.f, 0.f};
#pragma unroll 2
    for (int k0 = 0; k0 < HID; k0 += 32) {
      const bf16_t* ap = sh + lrow * HID + k0 + 2 * h8;
      const bf16_t* lp = sl + lrow * HID + k0 + 2 * h8;
      FragA ah, al;
      ah.h[0] = *(const v8bf*)(ap);
      ah.h[1] = *(const v8bf*)(ap + 8);
      al.h[0] = *(const v8bf*)(lp);
      al.h[1] = *(const v8bf*)(lp + 8);
      v16bf bhf = *(const v16bf*)(bh + k0 + 2 * h8);
      v16bf blf = *(const v16bf*)(bl + k0 + 2 * h8);
      acc = wmma_bf16(ah.v, bhf, acc);               // hi*hi
      acc = wmma_bf16(al.v, bhf, acc);               // lo*hi
      acc = wmma_bf16(ah.v, blf, acc);               // hi*lo
    }
#pragma unroll
    for (int v = 0; v < 8; ++v)
      m[(size_t)(row0 + chalf + v) * HID + ccol] = acc[v];
  }
}

// ---------------------------------------------------------------- fused GRU cell (split bf16 WMMA)
__global__ __launch_bounds__(256) void ggnn_gru(
    const float* __restrict__ agg, const float* __restrict__ x,
    const bf16_t* __restrict__ wih_h, const bf16_t* __restrict__ wih_l,
    const bf16_t* __restrict__ whh_h, const bf16_t* __restrict__ whh_l,
    const float* __restrict__ b_ih, const float* __restrict__ b_hh,
    float* __restrict__ out) {
  __shared__ bf16_t sAh[16 * HID];                   // agg hi  (16 KB)
  __shared__ bf16_t sAl[16 * HID];                   // agg lo
  __shared__ bf16_t sXh[16 * HID];                   // x hi
  __shared__ bf16_t sXl[16 * HID];                   // x lo   -> 64 KB total
  const int row0 = blockIdx.x * 16;
  const int tid = threadIdx.x;

  const float* asrc = agg + (size_t)row0 * HID;
  const float* xsrc = x + (size_t)row0 * HID;
  for (int i = tid; i < 16 * HID / 4; i += 256) {
    float4 va = ((const float4*)asrc)[i];
    float4 vx = ((const float4*)xsrc)[i];
    float fa[4] = {va.x, va.y, va.z, va.w};
    float fx[4] = {vx.x, vx.y, vx.z, vx.w};
    v4bf ah, al, xh, xl;
#pragma unroll
    for (int q = 0; q < 4; ++q) {
      bf16_t hb = (bf16_t)fa[q];
      ah[q] = hb;
      al[q] = (bf16_t)(fa[q] - (float)hb);
      bf16_t xb = (bf16_t)fx[q];
      xh[q] = xb;
      xl[q] = (bf16_t)(fx[q] - (float)xb);
    }
    *(v4bf*)(sAh + i * 4) = ah;
    *(v4bf*)(sAl + i * 4) = al;
    *(v4bf*)(sXh + i * 4) = xh;
    *(v4bf*)(sXl + i * 4) = xl;
  }
  __syncthreads();

  const int wave = tid >> 5, lane = tid & 31;
  const int lrow = lane & 15;
  const int h8 = (lane >> 4) << 3;
  const int chalf = (lane >> 4) << 3;

  for (int t = 0; t < 4; ++t) {
    const int col = wave * 64 + t * 16 + lrow;       // hidden unit [0,512)
    // B fragments: w[n][k] row-major, K contiguous per output column n.
    const bf16_t* wirh = wih_h + (size_t)col * HID;
    const bf16_t* wirl = wih_l + (size_t)col * HID;
    const bf16_t* wizh = wih_h + (size_t)(HID + col) * HID;
    const bf16_t* wizl = wih_l + (size_t)(HID + col) * HID;
    const bf16_t* winh = wih_h + (size_t)(2 * HID + col) * HID;
    const bf16_t* winl = wih_l + (size_t)(2 * HID + col) * HID;
    const bf16_t* whrh = whh_h + (size_t)col * HID;
    const bf16_t* whrl = whh_l + (size_t)col * HID;
    const bf16_t* whzh = whh_h + (size_t)(HID + col) * HID;
    const bf16_t* whzl = whh_l + (size_t)(HID + col) * HID;
    const bf16_t* whnh = whh_h + (size_t)(2 * HID + col) * HID;
    const bf16_t* whnl = whh_l + (size_t)(2 * HID + col) * HID;

    v8f racc = {0.f, 0.f, 0.f, 0.f, 0.f, 0.f, 0.f, 0.f};
    v8f zacc = racc, inacc = racc, hnacc = racc;

#pragma unroll 2
    for (int k0 = 0; k0 < HID; k0 += 32) {
      const int kb = k0 + 2 * h8;                    // B frag base (16 contiguous)
      const bf16_t* pAh = sAh + lrow * HID + kb;
      const bf16_t* pAl = sAl + lrow * HID + kb;
      const bf16_t* pXh = sXh + lrow * HID + kb;
      const bf16_t* pXl = sXl + lrow * HID + kb;
      FragA fah, fal, fxh, fxl;
      fah.h[0] = *(const v8bf*)(pAh); fah.h[1] = *(const v8bf*)(pAh + 8);
      fal.h[0] = *(const v8bf*)(pAl); fal.h[1] = *(const v8bf*)(pAl + 8);
      fxh.h[0] = *(const v8bf*)(pXh); fxh.h[1] = *(const v8bf*)(pXh + 8);
      fxl.h[0] = *(const v8bf*)(pXl); fxl.h[1] = *(const v8bf*)(pXl + 8);

      v16bf bh, bl;
      // r gate: agg@Wir + x@Whr
      bh = *(const v16bf*)(wirh + kb); bl = *(const v16bf*)(wirl + kb);
      racc = wmma_bf16(fah.v, bh, racc);
      racc = wmma_bf16(fal.v, bh, racc);
      racc = wmma_bf16(fah.v, bl, racc);
      bh = *(const v16bf*)(whrh + kb); bl = *(const v16bf*)(whrl + kb);
      racc = wmma_bf16(fxh.v, bh, racc);
      racc = wmma_bf16(fxl.v, bh, racc);
      racc = wmma_bf16(fxh.v, bl, racc);
      // z gate
      bh = *(const v16bf*)(wizh + kb); bl = *(const v16bf*)(wizl + kb);
      zacc = wmma_bf16(fah.v, bh, zacc);
      zacc = wmma_bf16(fal.v, bh, zacc);
      zacc = wmma_bf16(fah.v, bl, zacc);
      bh = *(const v16bf*)(whzh + kb); bl = *(const v16bf*)(whzl + kb);
      zacc = wmma_bf16(fxh.v, bh, zacc);
      zacc = wmma_bf16(fxl.v, bh, zacc);
      zacc = wmma_bf16(fxh.v, bl, zacc);
      // i_n (agg path)
      bh = *(const v16bf*)(winh + kb); bl = *(const v16bf*)(winl + kb);
      inacc = wmma_bf16(fah.v, bh, inacc);
      inacc = wmma_bf16(fal.v, bh, inacc);
      inacc = wmma_bf16(fah.v, bl, inacc);
      // h_n (x path)
      bh = *(const v16bf*)(whnh + kb); bl = *(const v16bf*)(whnl + kb);
      hnacc = wmma_bf16(fxh.v, bh, hnacc);
      hnacc = wmma_bf16(fxl.v, bh, hnacc);
      hnacc = wmma_bf16(fxh.v, bl, hnacc);
    }

    const float br  = b_ih[col] + b_hh[col];
    const float bz  = b_ih[HID + col] + b_hh[HID + col];
    const float bin = b_ih[2 * HID + col];
    const float bhn = b_hh[2 * HID + col];

#pragma unroll
    for (int v = 0; v < 8; ++v) {
      float r = sigmoidf_(racc[v] + br);
      float z = sigmoidf_(zacc[v] + bz);
      float n = tanhf(inacc[v] + bin + r * (hnacc[v] + bhn));
      float h = x[(size_t)(row0 + chalf + v) * HID + col];
      out[(size_t)(row0 + chalf + v) * HID + col] = (1.0f - z) * n + z * h;
    }
  }
}

// ----------------------------------------------------------------
extern "C" void kernel_launch(void* const* d_in, const int* in_sizes, int n_in,
                              void* d_out, int out_size, void* d_ws, size_t ws_size,
                              hipStream_t stream) {
  const int*   inputs = (const int*)d_in[0];
  const int*   A      = (const int*)d_in[1];
  const float* emb    = (const float*)d_in[2];
  const float* weight = (const float*)d_in[3];   // [2, 512, 512]
  const float* w_ih   = (const float*)d_in[4];   // [1536, 512]
  const float* w_hh   = (const float*)d_in[5];   // [1536, 512]
  const float* b_ih   = (const float*)d_in[6];
  const float* b_hh   = (const float*)d_in[7];
  float* out = (float*)d_out;

  const size_t NH = (size_t)NNODES * HID;        // 25.6M
  float* x   = (float*)d_ws;                     // [N, 512] fp32
  float* m   = x + NH;                           // [N, 512] fp32
  float* agg = m + NH;                           // [N, 512] fp32
  bf16_t* wth   = (bf16_t*)(agg + NH);           // [512, 512] Wt hi
  bf16_t* wtl   = wth + (size_t)HID * HID;       // Wt lo
  bf16_t* wih_h = wtl + (size_t)HID * HID;       // [1536, 512]
  bf16_t* wih_l = wih_h + (size_t)3 * HID * HID;
  bf16_t* whh_h = wih_l + (size_t)3 * HID * HID;
  bf16_t* whh_l = whh_h + (size_t)3 * HID * HID;

  const int n4 = NNODES * HID / 4;               // 6.4M float4s
  const int w4 = 3 * HID * HID / 4;              // 196608

  // one-time weight splits (deterministic, redone every call)
  ggnn_split<<<(w4 + 255) / 256, 256, 0, stream>>>(w_ih, wih_h, wih_l, w4);
  ggnn_split<<<(w4 + 255) / 256, 256, 0, stream>>>(w_hh, whh_h, whh_l, w4);

  ggnn_gather<<<(n4 + 255) / 256, 256, 0, stream>>>(inputs, emb, x);

  for (int l = 0; l < NLAYERS; ++l) {
    ggnn_tsplit_w<<<HID * HID / 256, 256, 0, stream>>>(
        weight + (size_t)l * HID * HID, wth, wtl);
    ggnn_gemm_bf16<<<NNODES / 16, 256, 0, stream>>>(x, wth, wtl, m);
    ggnn_zero<<<(n4 + 255) / 256, 256, 0, stream>>>(agg, n4);
    ggnn_scatter<<<(NEDGES * 128) / 256, 256, 0, stream>>>(A, m, agg);
    ggnn_gru<<<NNODES / 16, 256, 0, stream>>>(
        agg, x, wih_h, wih_l, whh_h, whh_l, b_ih, b_hh,
        (l == NLAYERS - 1) ? out : x);
  }
}
